// LSHAttn_56530359550025
// MI455X (gfx1250) — matrix operand order repaired and checked by reading, not compiled, not run on hardware
//
#include <hip/hip_runtime.h>
#include <hip/hip_bf16.h>
#include <cmath>

// ---------------------------------------------------------------------------
// LSH attention (Reformer) for MI455X / gfx1250, wave32.
//   B=32, T=4096, D=64, R=8 hash rounds, bucket=64, n_buckets=64, colors=128.
// Pipeline:
//   1) hash:   px = x @ proj[:,r,:]; bucket = argmax([px,-px])   (VALU)
//   2) sort:   stable counting sort of buckets per (b,r)          (LDS/DS)
//   3) attn:   per (b,r,bucket): gather->LDS f16, QK^T + PV via
//              v_wmma_f32_16x16x32_f16, register softmax (shfl_xor),
//              scatter output+logsumexp through sort permutation
//   4) combine rounds weighted by softmax of logsumexp over r
// Workspace need: 4*BRT*4 (buckets+stickers) + BRT*4 (logits)
//                 + BRT*64*4 (per-round o)  ~= 276 MB.
// ---------------------------------------------------------------------------

#define BB   32
#define TT   4096
#define DD   64
#define RR   8
#define NB   64      // n_buckets
#define BSZ  64      // bucket size

typedef __attribute__((ext_vector_type(16))) _Float16 v16h;
typedef __attribute__((ext_vector_type(8)))  _Float16 v8h;
typedef __attribute__((ext_vector_type(8)))  float    v8f;

static __device__ inline v16h ld16(const _Float16* p0, const _Float16* p1) {
  union { v16h v; v8h h[2]; } u;
  u.h[0] = *(const v8h*)p0;
  u.h[1] = *(const v8h*)p1;
  return u.v;
}

// ---------------------------------------------------------------------------
// Kernel 1: hashing. grid = 2*B*T blocks of 64 threads (q then k).
// ---------------------------------------------------------------------------
__global__ __launch_bounds__(64) void lsh_hash_kernel(
    const float* __restrict__ q, const float* __restrict__ k,
    const float* __restrict__ proj, int* __restrict__ bq, int* __restrict__ bk)
{
  const int BT = BB * TT;
  int id    = blockIdx.x;
  int which = (id >= BT);
  int bt    = which ? id - BT : id;
  const float* x = (which ? k : q) + (size_t)bt * DD;
  int* outb = which ? bk : bq;
  int b = bt >> 12, t = bt & (TT - 1);

  __shared__ float xs[DD];
  __shared__ float px[RR * 64];
  int n = threadIdx.x;              // 0..63 : color index within half
  xs[n] = x[n];
  __syncthreads();
  #pragma unroll
  for (int r = 0; r < RR; ++r) {
    float acc = 0.f;
    const float* pr = proj + r * 64 + n;   // proj[d][r][n] : d-stride 512
    #pragma unroll 8
    for (int d = 0; d < DD; ++d) acc = fmaf(xs[d], pr[d * (RR * 64)], acc);
    px[r * 64 + n] = acc;
  }
  __syncthreads();
  if (n < RR) {
    int r = n;
    float best = px[r * 64]; int bi = 0;
    for (int j = 1; j < 64; ++j) { float v0 = px[r * 64 + j]; if (v0 > best) { best = v0; bi = j; } }
    for (int j = 0; j < 64; ++j) { float v0 = -px[r * 64 + j]; if (v0 > best) { best = v0; bi = 64 + j; } }
    outb[(size_t)(b * RR + r) * TT + t] = bi;
  }
}

// ---------------------------------------------------------------------------
// Kernel 2: stable counting sort per (b,r). grid = 2*B*R blocks of 32 (1 wave).
// sticker[s] = original t of s-th element in stable (bucket,t) order.
// ---------------------------------------------------------------------------
__global__ __launch_bounds__(32) void lsh_sort_kernel(
    const int* __restrict__ bq, const int* __restrict__ bk,
    int* __restrict__ sq, int* __restrict__ sk)
{
  const int BR = BB * RR;           // 256
  int id = blockIdx.x;
  const int* buckets = (id >= BR) ? bk : bq;
  int*       sticker = (id >= BR) ? sk : sq;
  int br = (id >= BR) ? id - BR : id;
  size_t base = (size_t)br * TT;
  int lane = threadIdx.x;

  __shared__ int hist[128];
  __shared__ int sb[32];

  for (int i = lane; i < 128; i += 32) hist[i] = 0;
  __syncthreads();
  for (int g = 0; g < TT / 32; ++g)
    atomicAdd(&hist[buckets[base + g * 32 + lane]], 1);
  __syncthreads();
  if (lane == 0) { int run = 0; for (int i = 0; i < 128; ++i) { int c = hist[i]; hist[i] = run; run += c; } }
  __syncthreads();
  for (int g = 0; g < TT / 32; ++g) {
    int t = g * 32 + lane;
    int v = buckets[base + t];
    sb[lane] = v;
    __syncthreads();
    int eq = 0; bool last = true;
    #pragma unroll
    for (int j = 0; j < 32; ++j) {
      int w = sb[j];
      if (w == v) { if (j < lane) eq++; else if (j > lane) last = false; }
    }
    int pos = hist[v] + eq;
    sticker[base + pos] = t;
    __syncthreads();
    if (last) hist[v] = pos + 1;   // unique bucket per writer lane
    __syncthreads();
  }
}

// ---------------------------------------------------------------------------
// Kernel 3: bucketed attention. grid = B*R*NB blocks of 128 (4 waves).
// Each wave owns 16 query rows. WMMA f32<=f16 16x16x32.
// ---------------------------------------------------------------------------
__global__ __launch_bounds__(128) void lsh_attn_kernel(
    const float* __restrict__ q, const float* __restrict__ k,
    const float* __restrict__ v,
    const int* __restrict__ sq, const int* __restrict__ sk,
    float* __restrict__ o, float* __restrict__ logits)
{
  int nbk = blockIdx.x & (NB - 1);
  int r   = (blockIdx.x >> 6) & (RR - 1);
  int b   = blockIdx.x >> 9;
  size_t srt = (size_t)(b * RR + r) * TT;   // base into sticker/logits/o-rows
  int tid = threadIdx.x;

  // 56 KB static LDS
  __shared__ __align__(16) _Float16 smem[28672];
  _Float16* qs    = smem;              // 64 x 64   (8 KB)  queries (pre-scaled)
  _Float16* kt    = smem + 4096;       // 128 x 64  (16 KB) keys, row(key)-major
  _Float16* vts   = smem + 12288;      // 64 x 128  (16 KB) values, d-major (transposed)
  _Float16* Ptile = smem + 20480;      // 64 x 128  (16 KB) softmax probs
  float*    bo    = (float*)smem;      // 64 x 64 f32 overlays qs+kt after stage 1

  // ---- gather Q (scaled by d^-0.5), K, V^T into LDS as f16 ----
  {
    const float scale = 0.125f;        // 64^-0.5
    int row = tid >> 1;
    int c0  = (tid & 1) * 32;
    int tq  = sq[srt + nbk * BSZ + row];
    const float4* src = (const float4*)(q + ((size_t)b * TT + tq) * DD + c0);
    _Float16* dst = qs + row * DD + c0;
    #pragma unroll
    for (int j = 0; j < 8; ++j) {
      float4 f = src[j];
      dst[j*4+0] = (_Float16)(f.x * scale);
      dst[j*4+1] = (_Float16)(f.y * scale);
      dst[j*4+2] = (_Float16)(f.z * scale);
      dst[j*4+3] = (_Float16)(f.w * scale);
    }
  }
  {
    int i     = tid;                               // key slot 0..127
    int nprev = (nbk + NB - 1) & (NB - 1);         // look-one-back
    int sbuck = (i < 64) ? nprev : nbk;
    int tk    = sk[srt + sbuck * BSZ + (i & 63)];
    const float4* ksrc = (const float4*)(k + ((size_t)b * TT + tk) * DD);
    const float4* vsrc = (const float4*)(v + ((size_t)b * TT + tk) * DD);
    _Float16* kd = kt + i * DD;
    #pragma unroll
    for (int j = 0; j < 16; ++j) {
      float4 f = ksrc[j];
      kd[j*4+0] = (_Float16)f.x; kd[j*4+1] = (_Float16)f.y;
      kd[j*4+2] = (_Float16)f.z; kd[j*4+3] = (_Float16)f.w;
      float4 g = vsrc[j];
      vts[(j*4+0)*128 + i] = (_Float16)g.x;
      vts[(j*4+1)*128 + i] = (_Float16)g.y;
      vts[(j*4+2)*128 + i] = (_Float16)g.z;
      vts[(j*4+3)*128 + i] = (_Float16)g.w;
    }
  }
  __syncthreads();

  int lane  = tid & 31;
  int wv    = tid >> 5;
  int m0    = wv * 16;
  int lrow  = lane & 15;
  int lhalf = lane >> 4;
  int arow  = m0 + lrow;
  int c0    = lhalf * 8;

  // ---- stage 1: dots = Q*scale @ K^T, held entirely in VGPRs (8 tiles) ----
  v8f acc[8] = {};
  #pragma unroll
  for (int nt = 0; nt < 8; ++nt) {
    #pragma unroll
    for (int kk = 0; kk < 2; ++kk) {
      const _Float16* ap = qs + arow * DD + kk * 32 + c0;
      v16h A = ld16(ap, ap + 16);
      const _Float16* bp = kt + (nt * 16 + lrow) * DD + kk * 32 + lhalf * 8;
      v16h Bm = ld16(bp, bp + 16);
      acc[nt] = __builtin_amdgcn_wmma_f32_16x16x32_f16(
          false, A, false, Bm, (short)0, acc[nt], false, false);
    }
  }

  // ---- register softmax: row m = m0 + lhalf*8 + g lives on 16 lanes of one
  //      half-wave; xor masks 1/2/4/8 never cross the half boundary ----
  #pragma unroll
  for (int g = 0; g < 8; ++g) {
    float m = acc[0][g];
    #pragma unroll
    for (int nt = 1; nt < 8; ++nt) m = fmaxf(m, acc[nt][g]);
    #pragma unroll
    for (int msk = 8; msk >= 1; msk >>= 1) m = fmaxf(m, __shfl_xor(m, msk, 32));
    float s = 0.f;
    #pragma unroll
    for (int nt = 0; nt < 8; ++nt) s += __expf(acc[nt][g] - m);
    #pragma unroll
    for (int msk = 8; msk >= 1; msk >>= 1) s += __shfl_xor(s, msk, 32);
    float rs  = 1.0f / s;
    float lse = m + __logf(s);
    int row = m0 + lhalf * 8 + g;
    #pragma unroll
    for (int nt = 0; nt < 8; ++nt)
      Ptile[row * 128 + nt * 16 + lrow] = (_Float16)(__expf(acc[nt][g] - m) * rs);
    if (lrow == g) {                      // one lane per (half,g) -> one per row
      int tq = sq[srt + nbk * BSZ + row];
      logits[srt + tq] = lse;             // scatter == undo_sort gather
    }
  }
  __syncthreads();

  // ---- stage 2: bo = P @ V  (K = 128 keys, via d-major V^T tile) ----
  v8f acc2[4] = {};
  #pragma unroll
  for (int kk = 0; kk < 4; ++kk) {
    const _Float16* ap = Ptile + arow * 128 + kk * 32 + c0;
    v16h A = ld16(ap, ap + 16);
    #pragma unroll
    for (int nt = 0; nt < 4; ++nt) {
      const _Float16* bp = vts + (nt * 16 + lrow) * 128 + kk * 32 + lhalf * 8;
      v16h Bm = ld16(bp, bp + 16);
      acc2[nt] = __builtin_amdgcn_wmma_f32_16x16x32_f16(
          false, A, false, Bm, (short)0, acc2[nt], false, false);
    }
  }

  // stage-1 reads of qs/kt are behind the barrier above -> safe to overlay bo
  {
    int mbase = m0 + lhalf * 8;
    #pragma unroll
    for (int nt = 0; nt < 4; ++nt) {
      int col = nt * 16 + lrow;
      #pragma unroll
      for (int g = 0; g < 8; ++g) bo[(mbase + g) * DD + col] = acc2[nt][g];
    }
  }
  __syncthreads();

  // ---- scatter rows through sort permutation (== undo_sort) ----
  {
    int row = tid >> 1;
    int cc  = (tid & 1) * 32;
    int tq  = sq[srt + nbk * BSZ + row];
    float4* dst = (float4*)(o + ((size_t)(srt + tq)) * DD + cc);
    const float4* src = (const float4*)(bo + row * DD + cc);
    #pragma unroll
    for (int j = 0; j < 8; ++j) dst[j] = src[j];
  }
}

// ---------------------------------------------------------------------------
// Kernel 4: combine rounds. grid = B*T blocks of 64 threads (thread = d).
// ---------------------------------------------------------------------------
__global__ __launch_bounds__(64) void lsh_combine_kernel(
    const float* __restrict__ o, const float* __restrict__ logits,
    float* __restrict__ out)
{
  int bt = blockIdx.x;
  int b  = bt >> 12, t = bt & (TT - 1);
  int d  = threadIdx.x;
  __shared__ float lg[RR];
  if (d < RR) lg[d] = logits[(size_t)(b * RR + d) * TT + t];
  __syncthreads();
  float m = lg[0];
  #pragma unroll
  for (int rr = 1; rr < RR; ++rr) m = fmaxf(m, lg[rr]);
  float p[RR]; float s = 0.f;
  #pragma unroll
  for (int rr = 0; rr < RR; ++rr) { p[rr] = __expf(lg[rr] - m); s += p[rr]; }
  float rs = 1.0f / s;
  float acc = 0.f;
  #pragma unroll
  for (int rr = 0; rr < RR; ++rr)
    acc += o[((size_t)(b * RR + rr) * TT + t) * DD + d] * (p[rr] * rs);
  out[(size_t)bt * DD + d] = acc;
}

// ---------------------------------------------------------------------------
extern "C" void kernel_launch(void* const* d_in, const int* in_sizes, int n_in,
                              void* d_out, int out_size, void* d_ws, size_t ws_size,
                              hipStream_t stream) {
  (void)in_sizes; (void)n_in; (void)out_size; (void)ws_size;
  const float* q    = (const float*)d_in[0];
  const float* k    = (const float*)d_in[1];
  const float* v    = (const float*)d_in[2];
  const float* proj = (const float*)d_in[3];
  float* out = (float*)d_out;

  const size_t nBRT = (size_t)BB * RR * TT;   // 1,048,576
  int*   bq     = (int*)d_ws;
  int*   bk     = bq + nBRT;
  int*   sq     = bk + nBRT;
  int*   sk     = sq + nBRT;
  float* logits = (float*)(sk + nBRT);
  float* o      = logits + nBRT;              // B*R*T*D floats (256 MB)

  lsh_hash_kernel<<<2 * BB * TT, 64, 0, stream>>>(q, k, proj, bq, bk);
  lsh_sort_kernel<<<2 * BB * RR, 32, 0, stream>>>(bq, bk, sq, sk);
  lsh_attn_kernel<<<BB * RR * NB, 128, 0, stream>>>(q, k, v, sq, sk, o, logits);
  lsh_combine_kernel<<<BB * TT, 64, 0, stream>>>(o, logits, out);
}
